// BiLSTMModel_21440476742374
// MI455X (gfx1250) — matrix operand order repaired and checked by reading, not compiled, not run on hardware
//
#include <hip/hip_runtime.h>

typedef __attribute__((ext_vector_type(16))) _Float16 v16h;
typedef __attribute__((ext_vector_type(8)))  float    v8f;

#define HID 16
#define SEQ 512
#define INP 5

// fast sigmoid / tanh via native exp (v_exp_f32) + native rcp (v_rcp_f32)
__device__ __forceinline__ float fast_sig(float x) {
    return __builtin_amdgcn_rcpf(1.0f + __expf(-x));
}
__device__ __forceinline__ float fast_tanh(float x) {
    // tanh(x) = 2/(1+exp(-2x)) - 1
    return 2.0f * __builtin_amdgcn_rcpf(1.0f + __expf(-2.0f * x)) - 1.0f;
}

// One wave (32 lanes) per 16-batch tile; runs BOTH direction chains.
// GEMM orientation per gate block g:  D[16 gate-rows x 16 batch] =
//    A_g[16 x K=32] * B[K=32 x 16] + bias,  K = [h(16) | x(5) | pad(11)]
__global__ void __launch_bounds__(32)
bilstm_wmma_kernel(const float* __restrict__ x,
                   const float* __restrict__ Wih_f, const float* __restrict__ Whh_f,
                   const float* __restrict__ bih_f, const float* __restrict__ bhh_f,
                   const float* __restrict__ Wih_b, const float* __restrict__ Whh_b,
                   const float* __restrict__ bih_b, const float* __restrict__ bhh_b,
                   const float* __restrict__ Wfc,   const float* __restrict__ bfc,
                   float* __restrict__ out)
{
    const int lane = threadIdx.x;      // 0..31 (wave32)
    const int n    = lane & 15;        // batch-in-tile (also A row M)
    const int hi   = lane >> 4;        // half-wave select
    const long base = (long)blockIdx.x * 16;

    // ---------------- load weights into A-layout registers (once) ----------
    // A operand in this kernel's contiguous K order: lane L holds M=L%16;
    //   half j -> K = j + 16*hi
    v16h aF[4], aB[4];
    v8f  biasF[4], biasB[4];
#pragma unroll
    for (int g = 0; g < 4; ++g) {
        v16h af = {}, ab = {};
        const int row = g * 16 + n;    // gate row (PyTorch order i,f,g,o)
#pragma unroll
        for (int j = 0; j < 16; ++j) {
            const int K = j + 16 * hi;
            float vf = 0.0f, vb = 0.0f;
            if (K < HID)            { vf = Whh_f[row * HID + K];        vb = Whh_b[row * HID + K]; }
            else if (K - HID < INP) { vf = Wih_f[row * INP + (K - HID)]; vb = Wih_b[row * INP + (K - HID)]; }
            af[j] = (_Float16)vf;
            ab[j] = (_Float16)vb;
        }
        aF[g] = af; aB[g] = ab;
        v8f bf, bb;
#pragma unroll
        for (int r = 0; r < 8; ++r) {
            const int m = g * 16 + r + 8 * hi;   // C/D row M = r + 8*hi
            bf[r] = bih_f[m] + bhh_f[m];
            bb[r] = bih_b[m] + bhh_b[m];
        }
        biasF[g] = bf; biasB[g] = bb;
    }

    // ---------------- recurrent state in C/D layout -------------------------
    v8f hF = {}, cF = {}, hB = {}, cB = {};

    const float* xrow = x + (base + n) * (long)(SEQ * INP);

    // software-pipelined x fetch (fwd reads t, bwd reads SEQ-1-t)
    float xf[INP], xb[INP];
#pragma unroll
    for (int i = 0; i < INP; ++i) {
        xf[i] = xrow[i];
        xb[i] = xrow[(SEQ - 1) * INP + i];
    }

#pragma unroll 1
    for (int t = 0; t < SEQ; ++t) {
        // prefetch next timestep (clamped; unused values on last iter)
        const int tn = (t + 1 < SEQ) ? (t + 1) : t;
        float xfn[INP], xbn[INP];
        {
            const float* pf = xrow + (long)tn * INP;
            const float* pb = xrow + (long)(SEQ - 1 - tn) * INP;
#pragma unroll
            for (int i = 0; i < INP; ++i) { xfn[i] = pf[i]; xbn[i] = pb[i]; }
        }

        // -------- build fused B operand [h | x | 0] for both directions ----
        // B operand, same contiguous K order as A: lane N (0..15) holds K=0..15 (h rows),
        //                       lane N+16     holds K=16..31 (x rows + pad)
        float hoF[8], hoB[8];
#pragma unroll
        for (int r = 0; r < 8; ++r) {       // fetch hidden rows 8..15 from partner lane
            hoF[r] = __shfl_xor(hF[r], 16, 32);
            hoB[r] = __shfl_xor(hB[r], 16, 32);
        }
        v16h bFm, bBm;
#pragma unroll
        for (int r = 0; r < 8; ++r) {
            const float xFv = (r < INP) ? xf[r] : 0.0f;
            const float xBv = (r < INP) ? xb[r] : 0.0f;
            bFm[r]     = (_Float16)(hi ? xFv  : hF[r]);
            bFm[r + 8] = (_Float16)(hi ? 0.0f : hoF[r]);
            bBm[r]     = (_Float16)(hi ? xBv  : hB[r]);
            bBm[r + 8] = (_Float16)(hi ? 0.0f : hoB[r]);
        }

        // -------- 4 WMMAs per direction: all gate pre-activations ----------
        v8f giF = __builtin_amdgcn_wmma_f32_16x16x32_f16(false, aF[0], false, bFm, (short)0, biasF[0], false, false);
        v8f gfF = __builtin_amdgcn_wmma_f32_16x16x32_f16(false, aF[1], false, bFm, (short)0, biasF[1], false, false);
        v8f ggF = __builtin_amdgcn_wmma_f32_16x16x32_f16(false, aF[2], false, bFm, (short)0, biasF[2], false, false);
        v8f goF = __builtin_amdgcn_wmma_f32_16x16x32_f16(false, aF[3], false, bFm, (short)0, biasF[3], false, false);
        v8f giB = __builtin_amdgcn_wmma_f32_16x16x32_f16(false, aB[0], false, bBm, (short)0, biasB[0], false, false);
        v8f gfB = __builtin_amdgcn_wmma_f32_16x16x32_f16(false, aB[1], false, bBm, (short)0, biasB[1], false, false);
        v8f ggB = __builtin_amdgcn_wmma_f32_16x16x32_f16(false, aB[2], false, bBm, (short)0, biasB[2], false, false);
        v8f goB = __builtin_amdgcn_wmma_f32_16x16x32_f16(false, aB[3], false, bBm, (short)0, biasB[3], false, false);

        // -------- elementwise cell update (f32 state) -----------------------
#pragma unroll
        for (int r = 0; r < 8; ++r) {
            {
                const float iv = fast_sig(giF[r]);
                const float fv = fast_sig(gfF[r]);
                const float gv = fast_tanh(ggF[r]);
                const float ov = fast_sig(goF[r]);
                const float cn = fv * cF[r] + iv * gv;
                cF[r] = cn;
                hF[r] = ov * fast_tanh(cn);
            }
            {
                const float iv = fast_sig(giB[r]);
                const float fv = fast_sig(gfB[r]);
                const float gv = fast_tanh(ggB[r]);
                const float ov = fast_sig(goB[r]);
                const float cn = fv * cB[r] + iv * gv;
                cB[r] = cn;
                hB[r] = ov * fast_tanh(cn);
            }
        }

#pragma unroll
        for (int i = 0; i < INP; ++i) { xf[i] = xfn[i]; xb[i] = xbn[i]; }
    }

    // ---------------- final FC: out[b] = [hF|hB] . Wfc + bfc ----------------
    // lane (n,hi) holds h[r+8*hi][n] in reg r
    float acc = 0.0f;
#pragma unroll
    for (int r = 0; r < 8; ++r) {
        acc += Wfc[r + 8 * hi]       * hF[r];
        acc += Wfc[16 + r + 8 * hi]  * hB[r];
    }
    acc += __shfl_xor(acc, 16, 32);          // combine hidden 0..7 and 8..15
    if (hi == 0) out[base + n] = acc + bfc[0];
}

extern "C" void kernel_launch(void* const* d_in, const int* in_sizes, int n_in,
                              void* d_out, int out_size, void* d_ws, size_t ws_size,
                              hipStream_t stream) {
    const float* x     = (const float*)d_in[0];
    const float* Wih_f = (const float*)d_in[1];
    const float* Whh_f = (const float*)d_in[2];
    const float* bih_f = (const float*)d_in[3];
    const float* bhh_f = (const float*)d_in[4];
    const float* Wih_b = (const float*)d_in[5];
    const float* Whh_b = (const float*)d_in[6];
    const float* bih_b = (const float*)d_in[7];
    const float* bhh_b = (const float*)d_in[8];
    const float* Wfc   = (const float*)d_in[9];
    const float* bfc   = (const float*)d_in[10];

    const int B     = in_sizes[0] / (SEQ * INP);  // 8192
    const int tiles = B / 16;                     // 512 waves, one per tile

    bilstm_wmma_kernel<<<tiles, 32, 0, stream>>>(
        x, Wih_f, Whh_f, bih_f, bhh_f,
        Wih_b, Whh_b, bih_b, bhh_b,
        Wfc, bfc, (float*)d_out);
}
